// MQAAttention_58566174048665
// MI455X (gfx1250) — compile-verified, matrix operand order checked by reading, not run on hardware
//
#include <hip/hip_runtime.h>
#include <hip/hip_bf16.h>

typedef __bf16 bf16_t;
typedef __attribute__((ext_vector_type(16))) __bf16 v16bf;
typedef __attribute__((ext_vector_type(8)))  __bf16 v8bf;
typedef __attribute__((ext_vector_type(4)))  __bf16 v4bf;
typedef __attribute__((ext_vector_type(8)))  float  v8f;

// ---------------------------------------------------------------------------
// Fragment loader for 16-bit WMMA operands (A 16x32 or B 32x16 layouts).
// Per CDNA5 ISA: lane L holds row/col (L&15); low-half lanes hold K chunks
// {0..7, 16..23}, high-half lanes hold {8..15, 24..31}.  Caller passes the
// pointer already offset by (lane>>4)*8 along K; we fetch p[0..7], p[16..23].
// ---------------------------------------------------------------------------
__device__ __forceinline__ v16bf load_frag(const bf16_t* p) {
    union { v16bf v; v8bf h[2]; } u;
    u.h[0] = *(const v8bf*)(p);
    u.h[1] = *(const v8bf*)(p + 16);
    return u.v;
}

__device__ __forceinline__ v8f wmma_bf16(v16bf a, v16bf b, v8f c) {
    return __builtin_amdgcn_wmma_f32_16x16x32_bf16(
        /*neg_a=*/false, a, /*neg_b=*/false, b,
        /*c_mod=*/(short)0, c, /*reuse_a=*/false, /*reuse_b=*/false);
}

// ---------------------------------------------------------------------------
// fp32 -> bf16 conversion (4 elements / thread)
// ---------------------------------------------------------------------------
__global__ __launch_bounds__(256) void cvt_f32_bf16_x4(
    const float4* __restrict__ src, bf16_t* __restrict__ dst, int n4)
{
    int i = blockIdx.x * blockDim.x + threadIdx.x;
    if (i < n4) {
        float4 v = src[i];
        v4bf o;
        o[0] = (bf16_t)v.x; o[1] = (bf16_t)v.y;
        o[2] = (bf16_t)v.z; o[3] = (bf16_t)v.w;
        *(v4bf*)(dst + 4 * (size_t)i) = o;
    }
}

// ---------------------------------------------------------------------------
// Transpose V out of packed kv: kv (B*T, 2d) cols [d,2d) -> vt (B, d, T)
// ---------------------------------------------------------------------------
__global__ __launch_bounds__(256) void transpose_v(
    const bf16_t* __restrict__ kv, bf16_t* __restrict__ vt)
{
    constexpr int T = 2048, D = 128, LDKV = 256;
    int idx = blockIdx.x * blockDim.x + threadIdx.x;   // B*T*D total
    int dd = idx & (D - 1);
    int t  = (idx >> 7) & (T - 1);
    int b  = idx >> 18;
    vt[((size_t)b * D + dd) * T + t] =
        kv[((size_t)b * T + t) * LDKV + D + dd];
}

// ---------------------------------------------------------------------------
// GEMM:  out(M,N) = A(M,K) @ W(N,K)^T + bias(N)
// A, W bf16 row-major.  One wave -> 32x64 strip (2 M-subtiles x 4 N-subtiles,
// B-fragments reused across both M-subtiles: 12 b128 loads per 8 WMMAs).
// 8 waves/block -> 256x64 block tile.  Writes bf16 (out_bf) or fp32 (out_f).
// All dims assumed multiples of the tile sizes.
// ---------------------------------------------------------------------------
__global__ __launch_bounds__(256) void gemm_bf16_wmma(
    const bf16_t* __restrict__ A, const bf16_t* __restrict__ W,
    const float* __restrict__ bias,
    bf16_t* __restrict__ out_bf, float* __restrict__ out_f,
    int M, int N, int K, int lda, int ldb, int ldc)
{
    const int lane = threadIdx.x & 31;
    const int wave = threadIdx.x >> 5;
    const int half = lane >> 4;      // K-chunk selector
    const int lr   = lane & 15;      // row (A) / col (B,C) within tile
    const int m0   = blockIdx.x * 256 + wave * 32;
    const int n0   = blockIdx.y * 64;
    if (m0 >= M || n0 >= N) return;

    v8f acc[2][4] = {};
    const bf16_t* arow0 = A + (size_t)(m0 +  0 + lr) * lda + half * 8;
    const bf16_t* arow1 = A + (size_t)(m0 + 16 + lr) * lda + half * 8;
    const bf16_t* brow[4];
    #pragma unroll
    for (int j = 0; j < 4; ++j)
        brow[j] = W + (size_t)(n0 + j * 16 + lr) * ldb + half * 8;

    for (int k = 0; k < K; k += 32) {
        __builtin_prefetch(arow0 + k + 256, 0, 3);   // WGP-scope prefetch
        __builtin_prefetch(arow1 + k + 256, 0, 3);
        v16bf a0 = load_frag(arow0 + k);
        v16bf a1 = load_frag(arow1 + k);
        #pragma unroll
        for (int j = 0; j < 4; ++j) {
            v16bf b = load_frag(brow[j] + k);
            acc[0][j] = wmma_bf16(a0, b, acc[0][j]);
            acc[1][j] = wmma_bf16(a1, b, acc[1][j]);
        }
    }

    #pragma unroll
    for (int mi = 0; mi < 2; ++mi) {
        #pragma unroll
        for (int j = 0; j < 4; ++j) {
            const int col = n0 + j * 16 + lr;
            const float bv = bias[col];
            #pragma unroll
            for (int r = 0; r < 8; ++r) {
                const int row = m0 + mi * 16 + r + half * 8;   // C/D layout
                const float v = acc[mi][j][r] + bv;
                if (out_f) out_f[(size_t)row * ldc + col] = v;
                else       out_bf[(size_t)row * ldc + col] = (bf16_t)v;
            }
        }
    }
}

// ---------------------------------------------------------------------------
// Causal MQA flash attention.  One wave per (b, h, 16-row q tile).
// q  : (B*T, C) bf16, head h at cols [h*D, h*D+D)
// kv : (B*T, 2D) bf16, K at cols [0, D)
// vt : (B, D, T) bf16  (V transposed for contiguous-K B-fragments)
// y  : (B*T, C) bf16
// ---------------------------------------------------------------------------
__global__ __launch_bounds__(32) void mqa_flash_wmma(
    const bf16_t* __restrict__ q, const bf16_t* __restrict__ kv,
    const bf16_t* __restrict__ vt, bf16_t* __restrict__ y)
{
    constexpr int T = 2048, C = 2048, H = 16, D = 128, LDKV = 256;
    constexpr float SCALE = 0.08838834764831845f;   // 1/sqrt(128)

    __shared__ __attribute__((aligned(16))) bf16_t Pl[16 * 32];

    const int lane = threadIdx.x;
    const int half = lane >> 4;
    const int lr   = lane & 15;
    const int blk  = blockIdx.x;              // B * H * (T/16)
    const int t0   = (blk & (T / 16 - 1)) * 16;
    const int h    = (blk >> 7) & (H - 1);
    const int b    = blk >> 11;

    const bf16_t* qbase = q  + (size_t)(b * T + t0 + lr) * C + h * D + half * 8;
    const bf16_t* kbase = kv + (size_t)(b * T) * LDKV;
    const bf16_t* vtb   = vt + (size_t)b * D * T;

    v8f O[8] = {};
    float mrow[8], lrow[8];
    #pragma unroll
    for (int r = 0; r < 8; ++r) { mrow[r] = -__builtin_inff(); lrow[r] = 0.f; }

    const int sEnd = t0 + 16;                 // causal bound (exclusive)
    for (int s0 = 0; s0 < sEnd; s0 += 32) {
        // ---- scores S = q @ K^T   (M=16 t, N=32 s, inner D=128) ----
        v8f sc0 = {}, sc1 = {};
        #pragma unroll
        for (int kk = 0; kk < D; kk += 32) {
            v16bf aq = load_frag(qbase + kk);
            v16bf k0 = load_frag(kbase + (size_t)(s0 +  0 + lr) * LDKV + kk + half * 8);
            v16bf k1 = load_frag(kbase + (size_t)(s0 + 16 + lr) * LDKV + kk + half * 8);
            sc0 = wmma_bf16(aq, k0, sc0);
            sc1 = wmma_bf16(aq, k1, sc1);
        }

        // ---- causal mask + online softmax; write P (bf16) to LDS ----
        #pragma unroll
        for (int r = 0; r < 8; ++r) {
            const int t  = t0 + r + half * 8;          // this lane's row
            const int sa = s0 + lr;
            const int sb = s0 + 16 + lr;
            float a0 = (sa <= t) ? sc0[r] * SCALE : -__builtin_inff();
            float a1 = (sb <= t) ? sc1[r] * SCALE : -__builtin_inff();
            float mx = fmaxf(a0, a1);
            #pragma unroll
            for (int xm = 1; xm < 16; xm <<= 1)        // stays in 16-lane half
                mx = fmaxf(mx, __shfl_xor(mx, xm, 32));
            const float mnew = fmaxf(mrow[r], mx);
            const float p0 = __expf(a0 - mnew);
            const float p1 = __expf(a1 - mnew);
            const float corr = __expf(mrow[r] - mnew);
            float ps = p0 + p1;
            #pragma unroll
            for (int xm = 1; xm < 16; xm <<= 1)
                ps += __shfl_xor(ps, xm, 32);
            lrow[r] = lrow[r] * corr + ps;
            mrow[r] = mnew;
            const int m_ = r + half * 8;
            Pl[m_ * 32 + lr]      = (bf16_t)p0;
            Pl[m_ * 32 + 16 + lr] = (bf16_t)p1;
            #pragma unroll
            for (int j = 0; j < 8; ++j) O[j][r] *= corr;
        }
        asm volatile("s_wait_dscnt 0" ::: "memory");

        // ---- O += P @ V  (M=16 t, K=32 s, N=128 d) ----
        union { v16bf v; v8bf hh[2]; } up;
        up.hh[0] = *(const v8bf*)&Pl[lr * 32 + half * 8];
        up.hh[1] = *(const v8bf*)&Pl[lr * 32 + half * 8 + 16];
        const v16bf ap = up.v;
        #pragma unroll
        for (int nd = 0; nd < 8; ++nd) {
            const bf16_t* vp = vtb + (size_t)(nd * 16 + lr) * T + s0 + half * 8;
            v16bf bv = load_frag(vp);
            O[nd] = wmma_bf16(ap, bv, O[nd]);
        }
    }

    // ---- finalize: y = O / l ----
    #pragma unroll
    for (int r = 0; r < 8; ++r) {
        const float inv = 1.f / lrow[r];
        const int row = b * T + t0 + r + half * 8;
        #pragma unroll
        for (int nd = 0; nd < 8; ++nd)
            y[(size_t)row * C + h * D + nd * 16 + lr] = (bf16_t)(O[nd][r] * inv);
    }
}

// ---------------------------------------------------------------------------
extern "C" void kernel_launch(void* const* d_in, const int* in_sizes, int n_in,
                              void* d_out, int out_size, void* d_ws, size_t ws_size,
                              hipStream_t stream) {
    const float* x     = (const float*)d_in[0];
    const float* Wq    = (const float*)d_in[1];
    const float* bq    = (const float*)d_in[2];
    const float* Wkv   = (const float*)d_in[3];
    const float* bkv   = (const float*)d_in[4];
    const float* Wproj = (const float*)d_in[5];
    const float* bproj = (const float*)d_in[6];
    float* out = (float*)d_out;

    constexpr int B = 2, T = 2048, C = 2048, H = 16, D = 128;
    constexpr int M = B * T;            // 4096 token rows
    constexpr int KV = 2 * D;           // 256

    char* ws = (char*)d_ws;
    size_t off = 0;
    auto alloc = [&](size_t bytes) {
        void* p = ws + off;
        off += (bytes + 255) & ~size_t(255);
        return p;
    };
    bf16_t* xb   = (bf16_t*)alloc((size_t)M * C * 2);
    bf16_t* Wqb  = (bf16_t*)alloc((size_t)C * C * 2);
    bf16_t* Wkvb = (bf16_t*)alloc((size_t)KV * C * 2);
    bf16_t* Wpb  = (bf16_t*)alloc((size_t)C * C * 2);
    bf16_t* qb   = (bf16_t*)alloc((size_t)M * C * 2);
    bf16_t* kvb  = (bf16_t*)alloc((size_t)M * KV * 2);
    bf16_t* vtb  = (bf16_t*)alloc((size_t)B * D * T * 2);
    bf16_t* yb   = (bf16_t*)alloc((size_t)M * C * 2);

    auto cvt = [&](const float* s, bf16_t* d, size_t n) {
        int n4 = (int)(n / 4);
        cvt_f32_bf16_x4<<<(n4 + 255) / 256, 256, 0, stream>>>((const float4*)s, d, n4);
    };
    cvt(x,     xb,   (size_t)M * C);
    cvt(Wq,    Wqb,  (size_t)C * C);
    cvt(Wkv,   Wkvb, (size_t)KV * C);
    cvt(Wproj, Wpb,  (size_t)C * C);

    // q = x @ Wq^T + bq            (M x C)
    gemm_bf16_wmma<<<dim3(M / 256, C / 64), 256, 0, stream>>>(
        xb, Wqb, bq, qb, nullptr, M, C, C, C, C, C);
    // kv = x @ Wkv^T + bkv         (M x 2D)
    gemm_bf16_wmma<<<dim3(M / 256, KV / 64), 256, 0, stream>>>(
        xb, Wkvb, bkv, kvb, nullptr, M, KV, C, C, C, KV);
    // V^T staging                  (B, D, T)
    transpose_v<<<(B * T * D) / 256, 256, 0, stream>>>(kvb, vtb);
    // flash attention              y (M x C)
    mqa_flash_wmma<<<B * H * (T / 16), 32, 0, stream>>>(qb, kvb, vtb, yb);
    // out = y @ Wproj^T + bproj    (M x C), fp32 -> d_out
    gemm_bf16_wmma<<<dim3(M / 256, C / 64), 256, 0, stream>>>(
        yb, Wpb, bproj, nullptr, out, M, C, C, C, C, C);
}
